// LSTMModel_8143257993476
// MI455X (gfx1250) — matrix operand order repaired and checked by reading, not compiled, not run on hardware
//
#include <hip/hip_runtime.h>
#include <hip/hip_bf16.h>

// Problem constants (from reference): B=2048, T=1024, D=4, H=50
#define Tn      1024
#define Dn      4
#define Hn      50
#define MT      16          // batch rows per workgroup (WMMA M)
#define NTHREADS 128        // 4 waves (wave32)
#define XCHUNK  64          // timesteps of x staged per LDS chunk
#define ASTR    72          // A_sh row stride (bf16 elems) = 36 dwords -> conflict-free b128 loads
#define GSTR    260         // gate scratch row stride (dwords) -> conflict-free b32 stores
#define XSTR    260         // x staging row stride (dwords), 16B-aligned rows

#define LOG2E   1.4426950408889634f

typedef __attribute__((ext_vector_type(16))) __bf16 v16bf;
typedef __attribute__((ext_vector_type(8)))  __bf16 v8bf;
typedef __attribute__((ext_vector_type(8)))  float  v8f;

__device__ __forceinline__ unsigned short f2bf_bits(float f) {
    union { float f; unsigned u; } v; v.f = f;
    unsigned r = v.u + 0x7FFFu + ((v.u >> 16) & 1u);   // round-to-nearest-even
    return (unsigned short)(r >> 16);
}
__device__ __forceinline__ __bf16 bits2bf(unsigned short s) {
    union { unsigned short s; __bf16 b; } v; v.s = s; return v.b;
}
// fast tanh: 2*rcp(1+exp2(-2*log2e*x)) - 1   (v_exp_f32 + v_rcp_f32, no IEEE div)
__device__ __forceinline__ float tanh_fast(float x) {
    float e = __builtin_amdgcn_exp2f(-2.0f * LOG2E * x);
    return __builtin_fmaf(2.0f, __builtin_amdgcn_rcpf(1.0f + e), -1.0f);
}

// Extended-B element: rows 0..49 = W_hh^T, 50..53 = W_ih^T, 54 = fused bias, rest 0
__device__ __forceinline__ float bElem(const float* Whh, const float* Wih,
                                       const float* bih, const float* bhh,
                                       int k, int n, bool valid) {
    if (!valid) return 0.0f;
    if (k < Hn)        return Whh[n * Hn + k];
    if (k < Hn + Dn)   return Wih[n * Dn + (k - Hn)];
    if (k == Hn + Dn)  return bih[n] + bhh[n];
    return 0.0f;
}

__global__ __launch_bounds__(NTHREADS)
void lstm_fused_kernel(const float* __restrict__ x,   const float* __restrict__ Wih,
                       const float* __restrict__ Whh, const float* __restrict__ bih,
                       const float* __restrict__ bhh, const float* __restrict__ Wfc,
                       const float* __restrict__ bfc, float* __restrict__ out)
{
    __shared__ unsigned short A_sh[MT * ASTR];  // A operand: [h | x_t | 1 | 0] in bf16
    __shared__ float Gs[MT * GSTR];             // activated gates, groups at col w*64
    __shared__ float Cs[MT * 64];               // cell state (f32)
    __shared__ float XS[MT * XSTR];             // staged x chunk (f32)
    __shared__ float OutAcc[MT];

    const int tid  = threadIdx.x;
    const int lane = tid & 31;
    const int w    = tid >> 5;        // wave id = gate group (0:i 1:f 2:g 3:o)
    const int nl   = lane & 15;
    const int half = lane >> 4;
    const int bbase = blockIdx.x * MT;

    // Wave-uniform, branch-free activation constants:
    //   w!=2 (sigmoid): a = 1*rcp(1+exp2(-log2e*x)) + 0
    //   w==2 (tanh):    a = 2*rcp(1+exp2(-2*log2e*x)) - 1
    const float actK1 = (w == 2) ? (-2.0f * LOG2E) : (-LOG2E);
    const float actK2 = (w == 2) ? 2.0f : 1.0f;
    const float actK3 = (w == 2) ? -1.0f : 0.0f;

    // ---- init LDS state ----
    for (int i = tid; i < MT * ASTR; i += NTHREADS) A_sh[i] = 0;      // h0 = 0, pads = 0
    for (int i = tid; i < MT * 64;   i += NTHREADS) Cs[i] = 0.0f;     // c0 = 0
    if (tid < MT) OutAcc[tid] = bfc[0];
    __syncthreads();
    if (tid < MT) A_sh[tid * ASTR + (Hn + Dn)] = 0x3F80;              // bf16 1.0 (bias lane)

    // ---- loop-invariant B operands (extended weights) held in VGPRs ----
    v16bf Breg[4][2];
    #pragma unroll
    for (int nt = 0; nt < 4; ++nt) {
        const int  ncol  = nt * 16 + nl;
        const bool valid = ncol < Hn;
        const int  ng    = w * Hn + ncol;
        #pragma unroll
        for (int kt = 0; kt < 2; ++kt) {
            v16bf b;
            #pragma unroll
            for (int v = 0; v < 8; ++v) {
                int k0 = kt * 32 + half * 8 + 2 * v + (v >= 4 ? 8 : 0);   // B 32x16 table-order layout, pairs per VGPR
                b[2 * v]     = bits2bf(f2bf_bits(bElem(Whh, Wih, bih, bhh, k0,     ng, valid)));
                b[2 * v + 1] = bits2bf(f2bf_bits(bElem(Whh, Wih, bih, bhh, k0 + 1, ng, valid)));
            }
            Breg[nt][kt] = b;
        }
    }

    // ---- stage x chunk 0, write x_0 into A ----
    {
        int m = tid >> 3, sub = tid & 7;
        const float4* src = (const float4*)(x + ((size_t)(bbase + m) * Tn) * Dn) + sub * 8;
        float4* dst = (float4*)&XS[m * XSTR] + sub * 8;
        #pragma unroll
        for (int i = 0; i < 8; ++i) dst[i] = src[i];
        const char* p = (const char*)(x + ((size_t)(bbase + m) * Tn + XCHUNK) * Dn);
        __builtin_prefetch(p + sub * 128, 0, 1);        // global_prefetch_b8: next chunk
    }
    __syncthreads();
    if (tid < 64) {
        int m = tid >> 2, d = tid & 3;
        A_sh[m * ASTR + Hn + d] = f2bf_bits(XS[m * XSTR + d]);
    }
    __syncthreads();

    for (int t = 0; t < Tn; ++t) {
        // ---- gates = [h_t | x_t | 1] x [Whh^T ; Wih^T ; bias]  via bf16 WMMA ----
        const v8bf* arow = (const v8bf*)&A_sh[nl * ASTR + half * 8];  // ds_load_b128 x4
        v8bf a00 = arow[0];   // K-tile0: k = half*8 + 0..7
        v8bf a01 = arow[2];   //          k = half*8 + 16..23
        v8bf a10 = arow[4];   // K-tile1: +32
        v8bf a11 = arow[6];   //          +48
        v16bf A0 = __builtin_shufflevector(a00, a01, 0,1,2,3,4,5,6,7,8,9,10,11,12,13,14,15);
        v16bf A1 = __builtin_shufflevector(a10, a11, 0,1,2,3,4,5,6,7,8,9,10,11,12,13,14,15);

        #pragma unroll
        for (int nt = 0; nt < 4; ++nt) {
            v8f acc = {};
            acc = __builtin_amdgcn_wmma_f32_16x16x32_bf16(false, A0, false, Breg[nt][0],
                                                          (short)0, acc, false, false);
            acc = __builtin_amdgcn_wmma_f32_16x16x32_bf16(false, A1, false, Breg[nt][1],
                                                          (short)0, acc, false, false);
            #pragma unroll
            for (int v = 0; v < 8; ++v) {               // branch-free activation, spill
                float gv = acc[v];
                float e  = __builtin_amdgcn_exp2f(actK1 * gv);
                float av = __builtin_fmaf(actK2, __builtin_amdgcn_rcpf(1.0f + e), actK3);
                int m = v + half * 8;
                Gs[m * GSTR + w * 64 + nt * 16 + nl] = av;
            }
        }
        __syncthreads();

        // ---- elementwise c/h update across all 128 threads ----
        const bool last = (t == Tn - 1);
        for (int e = tid; e < MT * Hn; e += NTHREADS) {
            int m = e / Hn, j = e - m * Hn;
            float gi = Gs[m * GSTR +       j];
            float gf = Gs[m * GSTR +  64 + j];
            float gg = Gs[m * GSTR + 128 + j];
            float go = Gs[m * GSTR + 192 + j];
            float c  = __builtin_fmaf(gf, Cs[m * 64 + j], gi * gg);
            Cs[m * 64 + j] = c;
            float h = go * tanh_fast(c);
            A_sh[m * ASTR + j] = f2bf_bits(h);          // h_{t+1} as next A operand
            if (last) atomicAdd(&OutAcc[m], h * Wfc[j]); // fused final FC (ds_add_f32)
        }

        int tn = t + 1;
        if (tn < Tn) {
            if ((tn & (XCHUNK - 1)) == 0) {             // stage next x chunk
                int cidx = tn >> 6;
                int m = tid >> 3, sub = tid & 7;
                const float4* src = (const float4*)(x + ((size_t)(bbase + m) * Tn
                                       + (size_t)cidx * XCHUNK) * Dn) + sub * 8;
                float4* dst = (float4*)&XS[m * XSTR] + sub * 8;
                #pragma unroll
                for (int i = 0; i < 8; ++i) dst[i] = src[i];
                if (cidx + 1 < Tn / XCHUNK) {
                    const char* p = (const char*)(x + ((size_t)(bbase + m) * Tn
                                       + (size_t)(cidx + 1) * XCHUNK) * Dn);
                    __builtin_prefetch(p + sub * 128, 0, 1);
                }
                __syncthreads();                        // XS ready before x-write below
            }
            if (tid < 64) {                             // write x_{t+1} into A
                int m = tid >> 2, d = tid & 3, tc = tn & (XCHUNK - 1);
                A_sh[m * ASTR + Hn + d] = f2bf_bits(XS[m * XSTR + tc * 4 + d]);
            }
        }
        __syncthreads();                                // A complete, Gs reusable
    }

    if (tid < MT) out[bbase + tid] = OutAcc[tid];
}

extern "C" void kernel_launch(void* const* d_in, const int* in_sizes, int n_in,
                              void* d_out, int out_size, void* d_ws, size_t ws_size,
                              hipStream_t stream) {
    const float* x   = (const float*)d_in[0];
    const float* Wih = (const float*)d_in[1];
    const float* Whh = (const float*)d_in[2];
    const float* bih = (const float*)d_in[3];
    const float* bhh = (const float*)d_in[4];
    const float* Wfc = (const float*)d_in[5];
    const float* bfc = (const float*)d_in[6];
    float* out = (float*)d_out;

    const int Bsz = in_sizes[0] / (Tn * Dn);            // 2048
    dim3 grid(Bsz / MT), block(NTHREADS);
    lstm_fused_kernel<<<grid, block, 0, stream>>>(x, Wih, Whh, bih, bhh, Wfc, bfc, out);
}
